// StyledConv_455266533545
// MI455X (gfx1250) — matrix operand order repaired and checked by reading, not compiled, not run on hardware
//
#include <hip/hip_runtime.h>
#include <hip/hip_bf16.h>
#include <math.h>

typedef __attribute__((ext_vector_type(16))) __bf16 v16bf;
typedef __attribute__((ext_vector_type(8)))  float  v8f;

#define CIN  512
#define COUT 512
#define HW   4096
#define KD   4608   // 9 * 512
#define NKT  144    // KD / 32
#define PH   66     // padded height
#define PW   66     // padded width

// ---------------- prep: x NCHW f32 -> padded NHWC bf16 (zero halo) ----------------
__global__ void k_prep_x(const float* __restrict__ x, __bf16* __restrict__ xp) {
  size_t i = (size_t)blockIdx.x * 256 + threadIdx.x;   // 8*66*66*512 = 17,842,176
  int cin = (int)(i & 511);
  size_t t = i >> 9;
  int pw = (int)(t % PW); t /= PW;
  int ph = (int)(t % PH);
  int b  = (int)(t / PH);
  int h = ph - 1, w = pw - 1;
  float v = 0.f;
  if ((unsigned)h < 64u && (unsigned)w < 64u)
    v = x[((size_t)(b * CIN + cin)) * HW + h * 64 + w];
  xp[i] = (__bf16)v;
}

// ---- prep: kernel (k, cout) f32 -> B-fragment-packed bf16 ----
// layout: wf[kt][nt][lane][e], lane = n' + 16*(k'>>4), e = k'&15
__global__ void k_prep_w(const float* __restrict__ kern, __bf16* __restrict__ wf) {
  int idx  = blockIdx.x * 256 + threadIdx.x;           // < 144*32*32 = 147456
  int lane = idx & 31;
  int nt   = (idx >> 5) & 31;
  int kt   = idx >> 10;
  if (kt >= NKT) return;
  int cout = nt * 16 + (lane & 15);
  int kb   = kt * 32 + ((lane >> 4) << 4);
  __attribute__((aligned(16))) __bf16 tmp[16];
#pragma unroll
  for (int e = 0; e < 16; ++e)
    tmp[e] = (__bf16)kern[(size_t)(kb + e) * COUT + cout];
  float4* dst = (float4*)(wf + ((size_t)((kt * 32 + nt) * 32 + lane)) * 16);
  dst[0] = *(float4*)&tmp[0];
  dst[1] = *(float4*)&tmp[8];
}

// ---------------- prep: ksum[cout] = sum_k kernel^2 ----------------
__global__ void k_prep_ksum(const float* __restrict__ kern, float* __restrict__ ksum) {
  int c = blockIdx.x * 256 + threadIdx.x;
  if (c >= COUT) return;
  float s = 0.f;
  for (int k = 0; k < KD; ++k) { float v = kern[(size_t)k * COUT + c]; s = fmaf(v, v, s); }
  ksum[c] = s;
}

// ---- prep: g[b,cout] = (s+1) * rsqrt((s+1)^2 * ksum + eps) ----
__global__ void k_prep_g(const float* __restrict__ style, const float* __restrict__ ss,
                         const float* __restrict__ ksum, float* __restrict__ g) {
  int idx  = blockIdx.x * 256 + threadIdx.x;           // B*COUT = 4096
  int cout = idx & 511;
  int b    = idx >> 9;
  float s = 0.f;
  for (int i = 0; i < 512; ++i) s = fmaf(style[b * 512 + i], ss[i * 512 + cout], s);
  float m = s + 1.0f;
  g[idx] = m * rsqrtf(m * m * ksum[cout] + 1e-8f);
}

#define ASYNC_COPY_B128(ldsoff, gaddr)                                         \
  asm volatile("global_load_async_to_lds_b128 %0, %1, off"                     \
               :: "v"(ldsoff), "v"(gaddr) : "memory")
#define ASYNC_WAIT() asm volatile("s_wait_asynccnt 0x0" ::: "memory")

// ---------------- main conv: implicit GEMM, bf16 WMMA, async LDS staging ----------------
__global__ __launch_bounds__(256) void k_conv(
    const __bf16* __restrict__ xp, const __bf16* __restrict__ wf,
    const float* __restrict__ g,   const float* __restrict__ noise,
    const float* __restrict__ bias, const float* __restrict__ nwp,
    float* __restrict__ out)
{
  __shared__ __attribute__((aligned(32))) __bf16 ldsA[2][8 * 32 * 16];  // 8KB x2
  __shared__ __attribute__((aligned(32))) __bf16 ldsB[2][8 * 32 * 16];  // 8KB x2

  const int tid = threadIdx.x;
  const int b   = blockIdx.z;
  const int P0  = blockIdx.x * 128;   // pixel tile base
  const int N0  = blockIdx.y * 128;   // cout tile base
  const int nt0 = N0 >> 4;

  const int lane = tid & 31;
  const int wid  = tid >> 5;
  const int wp   = wid & 1;    // which 64-pixel half
  const int wn   = wid >> 1;   // which 32-cout quarter

  const __bf16* xpB = xp + (size_t)b * PH * PW * CIN;

  v8f zero = {};
  v8f acc[4][2];
#pragma unroll
  for (int i = 0; i < 4; ++i)
#pragma unroll
    for (int j = 0; j < 2; ++j) acc[i][j] = zero;

  auto stageA = [&](int buf, int kt) {
    int kbase = kt * 32;
    int r  = kbase >> 9;                 // 0..8 -> (kh,kw), pad already shifts by +1
    int dh = r / 3, dw = r % 3;
    int cin0 = kbase & 511;
#pragma unroll
    for (int u0 = 0; u0 < 2; ++u0) {
      int u  = tid * 2 + u0;             // 0..511
      int pp = u >> 2;                   // pixel within tile
      int gg = u & 3;                    // 8-wide k' group
      int p  = P0 + pp;
      int sh = (p >> 6) + dh;            // padded row
      int sw = (p & 63) + dw;            // padded col
      const __bf16* src = xpB + (size_t)(sh * PW + sw) * CIN + cin0 + gg * 8;
      int fm = pp >> 4, mp = pp & 15;
      unsigned loff = (unsigned)(uintptr_t)
          &ldsA[buf][(fm * 32 + mp + ((gg >> 1) << 4)) * 16 + ((gg & 1) << 3)];
      unsigned long long ga = (unsigned long long)(uintptr_t)src;
      ASYNC_COPY_B128(loff, ga);
    }
  };
  auto stageB = [&](int buf, int kt) {
    const __bf16* src = wf + ((size_t)(kt * 32 + nt0)) * 512;  // 8KB contiguous
#pragma unroll
    for (int q = 0; q < 2; ++q) {
      int vv = tid + q * 256;            // 0..511 chunks of 8 bf16
      unsigned loff = (unsigned)(uintptr_t)&ldsB[buf][vv * 8];
      unsigned long long ga = (unsigned long long)(uintptr_t)(src + vv * 8);
      ASYNC_COPY_B128(loff, ga);
    }
  };

  stageA(0, 0);
  stageB(0, 0);
  ASYNC_WAIT();
  __syncthreads();

  for (int kt = 0; kt < NKT; ++kt) {
    int cur = kt & 1;
    if (kt + 1 < NKT) { stageA(cur ^ 1, kt + 1); stageB(cur ^ 1, kt + 1); }

    v16bf afr[4], bfr[2];
#pragma unroll
    for (int i = 0; i < 4; ++i)
      afr[i] = *(const v16bf*)(&ldsA[cur][((wp * 4 + i) * 32 + lane) * 16]);
#pragma unroll
    for (int j = 0; j < 2; ++j)
      bfr[j] = *(const v16bf*)(&ldsB[cur][((wn * 2 + j) * 32 + lane) * 16]);

#pragma unroll
    for (int i = 0; i < 4; ++i)
#pragma unroll
      for (int j = 0; j < 2; ++j)
        acc[i][j] = __builtin_amdgcn_wmma_f32_16x16x32_bf16(
            false, afr[i], false, bfr[j], (short)0, acc[i][j], false, false);

    ASYNC_WAIT();          // async DMA for next buffer done (overlapped with WMMA)
    __syncthreads();
  }

  // epilogue: out = leaky(g*acc + nw*noise + bias) * sqrt(2)
  const float nw = nwp[0];
  const float SC = 1.41421356237309515f;
#pragma unroll
  for (int j = 0; j < 2; ++j) {
    int cout = N0 + wn * 32 + j * 16 + (lane & 15);
    float gv = g[b * COUT + cout];
    float bv = bias[cout];
    float* outc = out + ((size_t)b * COUT + cout) * HW;
#pragma unroll
    for (int i = 0; i < 4; ++i) {
#pragma unroll
      for (int r = 0; r < 8; ++r) {
        int pixel = P0 + wp * 64 + i * 16 + r + ((lane >> 4) << 3);
        float v = acc[i][j][r] * gv + nw * noise[(size_t)b * HW + pixel] + bv;
        v = (v > 0.f ? v : 0.2f * v) * SC;
        outc[pixel] = v;
      }
    }
  }
}

extern "C" void kernel_launch(void* const* d_in, const int* in_sizes, int n_in,
                              void* d_out, int out_size, void* d_ws, size_t ws_size,
                              hipStream_t stream) {
  const float* x     = (const float*)d_in[0];
  const float* style = (const float*)d_in[1];
  const float* noise = (const float*)d_in[2];
  const float* kern  = (const float*)d_in[3];
  const float* ss    = (const float*)d_in[4];
  const float* bias  = (const float*)d_in[5];
  const float* nw    = (const float*)d_in[6];
  float* out = (float*)d_out;

  char* ws = (char*)d_ws;
  __bf16* xp = (__bf16*)ws;
  size_t off = (size_t)8 * PH * PW * CIN * 2;                    // ~35.7 MB padded x
  __bf16* wf = (__bf16*)(ws + off); off += (size_t)NKT * 32 * 32 * 16 * 2;  // 4.5 MB
  float* ksum = (float*)(ws + off); off += 512 * 4;
  float* g    = (float*)(ws + off); off += 8 * COUT * 4;

  k_prep_x<<<(8 * PH * PW * CIN) / 256, 256, 0, stream>>>(x, xp);
  k_prep_w<<<576, 256, 0, stream>>>(kern, wf);
  k_prep_ksum<<<2, 256, 0, stream>>>(kern, ksum);
  k_prep_g<<<16, 256, 0, stream>>>(style, ss, ksum, g);

  dim3 grid(HW / 128, COUT / 128, 8);   // (32, 4, 8)
  k_conv<<<grid, 256, 0, stream>>>(xp, wf, g, noise, bias, nw, out);
}
